// CausalAttention_65403761983810
// MI455X (gfx1250) — matrix operand order repaired and checked by reading, not compiled, not run on hardware
//
#include <hip/hip_runtime.h>

#define DIM       1024
#define HEADS     8
#define DH        64
#define NCTX      4096
#define BATCH     2
#define BH        (BATCH * HEADS)
#define QKV_COLS  1536
#define ROWS      8192

// ---- Tensor Data Mover availability (arity differs across toolchains) ----
#if defined(__has_builtin)
#if __has_builtin(__builtin_amdgcn_tensor_load_to_lds) && \
    __has_builtin(__builtin_amdgcn_s_wait_tensorcnt)
#define HAVE_TDM 1
#endif
#endif
#ifndef HAVE_TDM
#define HAVE_TDM 0
#endif

typedef __attribute__((ext_vector_type(16))) __bf16 v16bf;
typedef __attribute__((ext_vector_type(8)))  __bf16 bf16x8;
typedef __attribute__((ext_vector_type(4)))  __bf16 bf16x4;
typedef __attribute__((ext_vector_type(8)))  float  v8f;
typedef __attribute__((ext_vector_type(4)))  float  f32x4;
typedef __attribute__((ext_vector_type(4)))  unsigned u32x4;
typedef __attribute__((ext_vector_type(8)))  int      i32x8;
typedef __attribute__((ext_vector_type(4)))  int      i32x4;

#if HAVE_TDM
// 2D tile load via TDM: tensor==tile (n0 x n1 elements of 2 bytes), row stride
// stride0 elements, LDS rows padded by 16B (pad_amount=4 dwords) after every
// row (pad_iv: 3 -> 16 dwords/row (n0=32), 4 -> 32 dwords/row (n0=64)).
__device__ __forceinline__ void tdm_load_2d(unsigned lds_off, const void* gptr,
                                            unsigned n0, unsigned n1,
                                            unsigned stride0, unsigned pad_iv) {
  const unsigned long long ga = (unsigned long long)(uintptr_t)gptr;
  u32x4 g0;
  g0[0] = 1u;                                                  // count=1
  g0[1] = lds_off;                                             // lds_addr
  g0[2] = (unsigned)ga;                                        // global[31:0]
  g0[3] = ((unsigned)(ga >> 32) & 0x01FFFFFFu) | (2u << 30);   // [56:32]|type=2
  i32x8 g1;
  g1[0] = (int)(0x00010000u | (1u << 20) | (pad_iv << 22) | (3u << 25));
  g1[1] = (int)((n0 & 0xFFFFu) << 16);                         // tensor_dim0 lo
  g1[2] = (int)(((n0 >> 16) & 0xFFFFu) | ((n1 & 0xFFFFu) << 16));
  g1[3] = (int)(((n1 >> 16) & 0xFFFFu) | ((n0 & 0xFFFFu) << 16)); // tile_dim0
  g1[4] = (int)(n1 & 0xFFFFu);                                 // tile_dim1
  g1[5] = (int)stride0;                                        // dim0 stride
  g1[6] = 0;
  g1[7] = 0;
  const i32x4 z4 = {0, 0, 0, 0};
#if __clang_major__ >= 23
  const i32x8 z8 = {0, 0, 0, 0, 0, 0, 0, 0};
  __builtin_amdgcn_tensor_load_to_lds(g0, g1, z4, z4, z8, 0);
#else
  __builtin_amdgcn_tensor_load_to_lds(g0, g1, z4, z4, 0);
#endif
}
#endif

// ---------------------------------------------------------------------------
// Kernel 1: RMSNorm (L2-normalize * sqrt(dim) * gamma), fp32 -> bf16
// ---------------------------------------------------------------------------
__global__ __launch_bounds__(256) void rmsnorm_kernel(
    const float* __restrict__ x, const float* __restrict__ gamma,
    __bf16* __restrict__ xn) {
  const int row = blockIdx.x;
  const int t   = threadIdx.x;
  const float* xr = x + (size_t)row * DIM;
  f32x4 v = *(const f32x4*)(xr + t * 4);
  float ss = v.x * v.x + v.y * v.y + v.z * v.z + v.w * v.w;
#pragma unroll
  for (int m = 16; m >= 1; m >>= 1) ss += __shfl_xor(ss, m, 32);
  __shared__ float warp_ss[8];
  if ((t & 31) == 0) warp_ss[t >> 5] = ss;
  __syncthreads();
  float tot = 0.f;
#pragma unroll
  for (int i = 0; i < 8; ++i) tot += warp_ss[i];
  float l2 = fmaxf(sqrtf(tot), 1e-12f);
  float scale = 32.0f / l2;  // sqrt(1024)
  f32x4 g = *(const f32x4*)(gamma + t * 4);
  bf16x4 o;
  o[0] = (__bf16)(v.x * scale * g.x);
  o[1] = (__bf16)(v.y * scale * g.y);
  o[2] = (__bf16)(v.z * scale * g.z);
  o[3] = (__bf16)(v.w * scale * g.w);
  *(bf16x4*)(xn + (size_t)row * DIM + t * 4) = o;
}

// ---------------------------------------------------------------------------
// Kernel 2: W [1024,1536] fp32 -> Wt [1536,1024] bf16 (tiled transpose)
// ---------------------------------------------------------------------------
__global__ __launch_bounds__(256) void wt_transpose_kernel(
    const float* __restrict__ w, __bf16* __restrict__ wt) {
  __shared__ float tile[32][33];
  const int c0 = blockIdx.x * 32;   // column of W
  const int k0 = blockIdx.y * 32;   // row of W
  const int tx = threadIdx.x & 31;
  const int ty = threadIdx.x >> 5;  // 0..7
#pragma unroll
  for (int i = 0; i < 4; ++i)
    tile[ty + i * 8][tx] = w[(size_t)(k0 + ty + i * 8) * QKV_COLS + c0 + tx];
  __syncthreads();
#pragma unroll
  for (int i = 0; i < 4; ++i)
    wt[(size_t)(c0 + ty + i * 8) * DIM + k0 + tx] = (__bf16)tile[tx][ty + i * 8];
}

// ---------------------------------------------------------------------------
// Kernel 3: QKV GEMM  C[8192,1536] = xn @ Wt^T   (both bf16)
// block tile 128x128, 8 waves of 32x64 (2x4 WMMA tiles), K-step 32,
// double-buffered LDS staged by TDM (fallback: vector copies)
// ---------------------------------------------------------------------------
__global__ __launch_bounds__(256) void qkv_gemm_kernel(
    const __bf16* __restrict__ xn, const __bf16* __restrict__ wt,
    __bf16* __restrict__ qb, __bf16* __restrict__ kbuf,
    __bf16* __restrict__ vtb) {
  __shared__ __align__(16) __bf16 As[2][128][40];  // [m][k] padded (+16B/row)
  __shared__ __align__(16) __bf16 Bs[2][128][40];  // [n][k] padded

  const int tid  = threadIdx.x;
  const int lane = tid & 31;
  const int wave = tid >> 5;
  const int wm   = wave >> 1;  // 0..3
  const int wn   = wave & 1;   // 0..1
  const int r0   = blockIdx.y * 128;
  const int c0   = blockIdx.x * 128;
  const int l15  = lane & 15;

  v8f zero = {0.f, 0.f, 0.f, 0.f, 0.f, 0.f, 0.f, 0.f};
  v8f acc[2][4];
#pragma unroll
  for (int mt = 0; mt < 2; ++mt)
#pragma unroll
    for (int nt = 0; nt < 4; ++nt) acc[mt][nt] = zero;

  const __bf16* Abase = xn + (size_t)r0 * DIM;
  const __bf16* Bbase = wt + (size_t)c0 * DIM;
  const int kbA = (lane >> 4) * 8;
  const int kbB = (lane >> 4) * 8;

#if HAVE_TDM
  if (wave == 0) {
    tdm_load_2d((unsigned)(uintptr_t)&As[0][0][0], Abase, 32, 128, DIM, 3);
    tdm_load_2d((unsigned)(uintptr_t)&Bs[0][0][0], Bbase, 32, 128, DIM, 3);
  }
#endif

  for (int kt = 0, it = 0; kt < DIM; kt += 32, ++it) {
    const int buf = it & 1;
#if HAVE_TDM
    if (wave == 0) {
      if (kt + 32 < DIM) {  // prefetch next tile pair into other buffer
        tdm_load_2d((unsigned)(uintptr_t)&As[buf ^ 1][0][0], Abase + kt + 32,
                    32, 128, DIM, 3);
        tdm_load_2d((unsigned)(uintptr_t)&Bs[buf ^ 1][0][0], Bbase + kt + 32,
                    32, 128, DIM, 3);
        __builtin_amdgcn_s_wait_tensorcnt(2);  // current buffer complete
      } else {
        __builtin_amdgcn_s_wait_tensorcnt(0);
      }
    }
    __syncthreads();
#else
    {
      const int row = tid >> 1, koff = (tid & 1) * 16;
      const __bf16* as = Abase + (size_t)row * DIM + kt + koff;
      *(bf16x8*)(&As[buf][row][koff])     = *(const bf16x8*)as;
      *(bf16x8*)(&As[buf][row][koff + 8]) = *(const bf16x8*)(as + 8);
      const __bf16* bs = Bbase + (size_t)row * DIM + kt + koff;
      *(bf16x8*)(&Bs[buf][row][koff])     = *(const bf16x8*)bs;
      *(bf16x8*)(&Bs[buf][row][koff + 8]) = *(const bf16x8*)(bs + 8);
    }
    __syncthreads();
#endif

    v16bf afrag[2], bfrag[4];
#pragma unroll
    for (int mt = 0; mt < 2; ++mt) {
      const __bf16* ap = &As[buf][wm * 32 + mt * 16 + l15][kbA];
      bf16x8 lo = *(const bf16x8*)ap;
      bf16x8 hi = *(const bf16x8*)(ap + 16);
#pragma unroll
      for (int i = 0; i < 8; ++i) { afrag[mt][i] = lo[i]; afrag[mt][i + 8] = hi[i]; }
    }
#pragma unroll
    for (int nt = 0; nt < 4; ++nt) {
      const __bf16* bp = &Bs[buf][wn * 64 + nt * 16 + l15][kbB];
      bf16x8 lo = *(const bf16x8*)bp;
      bf16x8 hi = *(const bf16x8*)(bp + 16);
#pragma unroll
      for (int i = 0; i < 8; ++i) { bfrag[nt][i] = lo[i]; bfrag[nt][i + 8] = hi[i]; }
    }
#pragma unroll
    for (int mt = 0; mt < 2; ++mt)
#pragma unroll
      for (int nt = 0; nt < 4; ++nt)
        acc[mt][nt] = __builtin_amdgcn_wmma_f32_16x16x32_bf16(
            false, afrag[mt], false, bfrag[nt], (short)0, acc[mt][nt],
            false, false);
    __syncthreads();
  }

  // epilogue: q scaled by dim^-0.5; v stored TRANSPOSED as [b,h,dh,n]
  const int hh = ((lane >> 4) & 1) * 8;
#pragma unroll
  for (int mt = 0; mt < 2; ++mt) {
#pragma unroll
    for (int nt = 0; nt < 4; ++nt) {
      const int gc     = c0 + wn * 64 + nt * 16 + l15;
      const int which  = gc >> 9;        // 0=q 1=k 2=v
      const int within = gc & 511;
      const int head   = within >> 6;
      const int dh     = within & 63;
#pragma unroll
      for (int e = 0; e < 8; ++e) {
        const int gr = r0 + wm * 32 + mt * 16 + hh + e;
        const int bb = gr >> 12;
        const int n  = gr & 4095;
        const int bhh = bb * HEADS + head;
        const float val = acc[mt][nt][e];
        if (which == 0)
          qb[((size_t)bhh * NCTX + n) * DH + dh] = (__bf16)(val * 0.03125f);
        else if (which == 1)
          kbuf[((size_t)bhh * NCTX + n) * DH + dh] = (__bf16)val;
        else
          vtb[((size_t)bhh * DH + dh) * NCTX + n] = (__bf16)val;
      }
    }
  }
}

// ---------------------------------------------------------------------------
// Kernel 4: causal flash attention; 4 waves x 16 Q rows; 64-key KV blocks,
// double-buffered K / V^T tiles via TDM (fallback: vector copies)
// ---------------------------------------------------------------------------
__global__ __launch_bounds__(128) void flash_attn_kernel(
    const __bf16* __restrict__ q, const __bf16* __restrict__ k,
    const __bf16* __restrict__ vt, float* __restrict__ out) {
  __shared__ __align__(16) __bf16 Ks[2][64][72];   // [key][dh] padded
  __shared__ __align__(16) __bf16 Vs[2][64][72];   // [dh][key] padded
  __shared__ __align__(16) __bf16 Ps[4][16][72];   // per-wave P scratch

  const int bh   = blockIdx.y;
  const int q0   = blockIdx.x * 64;
  const int tid  = threadIdx.x;
  const int lane = tid & 31;
  const int wave = tid >> 5;
  const int r0   = q0 + wave * 16;
  const int l15  = lane & 15;

  const __bf16* Qp  = q  + (size_t)bh * NCTX * DH;
  const __bf16* Kp  = k  + (size_t)bh * NCTX * DH;
  const __bf16* Vtp = vt + (size_t)bh * DH * NCTX;

  const int kbA = (lane >> 4) * 8;
  const int kbB = (lane >> 4) * 8;
  const int hh  = ((lane >> 4) & 1) * 8;

  // Q A-fragments (K dim 64 -> two 16x32 fragments), loaded once
  v16bf qa[2];
  {
    const int qrow = r0 + l15;
#pragma unroll
    for (int f = 0; f < 2; ++f) {
      const __bf16* src = Qp + (size_t)qrow * DH + f * 32 + kbA;
      bf16x8 lo = *(const bf16x8*)src;
      bf16x8 hi = *(const bf16x8*)(src + 16);
#pragma unroll
      for (int i = 0; i < 8; ++i) { qa[f][i] = lo[i]; qa[f][i + 8] = hi[i]; }
    }
  }

  float mrow[8], lrow[8], oacc[4][8];
#pragma unroll
  for (int e = 0; e < 8; ++e) {
    mrow[e] = -3.0e38f;
    lrow[e] = 0.f;
#pragma unroll
    for (int t = 0; t < 4; ++t) oacc[t][e] = 0.f;
  }

  const int jend = blockIdx.x;  // last 64-key block (kv0 <= q0)

#if HAVE_TDM
  if (wave == 0) {
    tdm_load_2d((unsigned)(uintptr_t)&Ks[0][0][0], Kp, 64, 64, DH, 4);
    tdm_load_2d((unsigned)(uintptr_t)&Vs[0][0][0], Vtp, 64, 64, NCTX, 4);
  }
#endif

  for (int j = 0; j <= jend; ++j) {
    const int kv0 = j * 64;
    const int buf = j & 1;
#if HAVE_TDM
    if (wave == 0) {
      if (j < jend) {
        tdm_load_2d((unsigned)(uintptr_t)&Ks[buf ^ 1][0][0],
                    Kp + (size_t)(kv0 + 64) * DH, 64, 64, DH, 4);
        tdm_load_2d((unsigned)(uintptr_t)&Vs[buf ^ 1][0][0],
                    Vtp + kv0 + 64, 64, 64, NCTX, 4);
        __builtin_amdgcn_s_wait_tensorcnt(2);
      } else {
        __builtin_amdgcn_s_wait_tensorcnt(0);
      }
    }
    __syncthreads();
#else
    {
      const int rr = tid >> 1, off = (tid & 1) * 32;
      const __bf16* ks = Kp + (size_t)(kv0 + rr) * DH + off;
#pragma unroll
      for (int i = 0; i < 4; ++i)
        *(bf16x8*)(&Ks[buf][rr][off + i * 8]) = *(const bf16x8*)(ks + i * 8);
      const __bf16* vs = Vtp + (size_t)rr * NCTX + kv0 + off;
#pragma unroll
      for (int i = 0; i < 4; ++i)
        *(bf16x8*)(&Vs[buf][rr][off + i * 8]) = *(const bf16x8*)(vs + i * 8);
    }
    __syncthreads();
#endif

    // ---- S = Q @ K^T : four 16x16 tiles x (K=64 -> 2 WMMAs) ----
    v8f s[4];
#pragma unroll
    for (int nt = 0; nt < 4; ++nt) {
      v8f z = {0.f, 0.f, 0.f, 0.f, 0.f, 0.f, 0.f, 0.f};
      s[nt] = z;
#pragma unroll
      for (int f = 0; f < 2; ++f) {
        v16bf bk;
        const __bf16* kp0 = &Ks[buf][nt * 16 + l15][f * 32 + kbB];
        bf16x8 lo = *(const bf16x8*)kp0;
        bf16x8 hi = *(const bf16x8*)(kp0 + 16);
#pragma unroll
        for (int i = 0; i < 8; ++i) { bk[i] = lo[i]; bk[i + 8] = hi[i]; }
        s[nt] = __builtin_amdgcn_wmma_f32_16x16x32_bf16(
            false, qa[f], false, bk, (short)0, s[nt], false, false);
      }
    }

    // ---- causal mask + online softmax ----
    float p[4][8];
#pragma unroll
    for (int e = 0; e < 8; ++e) {
      const int row = r0 + hh + e;
      float sv0 = s[0][e]; if (kv0 +  0 + l15 > row) sv0 = -1e30f;
      float sv1 = s[1][e]; if (kv0 + 16 + l15 > row) sv1 = -1e30f;
      float sv2 = s[2][e]; if (kv0 + 32 + l15 > row) sv2 = -1e30f;
      float sv3 = s[3][e]; if (kv0 + 48 + l15 > row) sv3 = -1e30f;
      float rm = fmaxf(fmaxf(sv0, sv1), fmaxf(sv2, sv3));
#pragma unroll
      for (int d = 1; d < 16; d <<= 1) rm = fmaxf(rm, __shfl_xor(rm, d, 32));
      const float mn    = fmaxf(mrow[e], rm);
      const float alpha = __expf(mrow[e] - mn);
      mrow[e] = mn;
      const float p0 = __expf(sv0 - mn);
      const float p1 = __expf(sv1 - mn);
      const float p2 = __expf(sv2 - mn);
      const float p3 = __expf(sv3 - mn);
      float rs = (p0 + p1) + (p2 + p3);
#pragma unroll
      for (int d = 1; d < 16; d <<= 1) rs += __shfl_xor(rs, d, 32);
      lrow[e] = lrow[e] * alpha + rs;
#pragma unroll
      for (int t = 0; t < 4; ++t) oacc[t][e] *= alpha;
      p[0][e] = p0; p[1][e] = p1; p[2][e] = p2; p[3][e] = p3;
    }

    // ---- C-layout -> A-layout via per-wave LDS scratch ----
#pragma unroll
    for (int e = 0; e < 8; ++e) {
      const int r = hh + e;
      Ps[wave][r][ 0 + l15] = (__bf16)p[0][e];
      Ps[wave][r][16 + l15] = (__bf16)p[1][e];
      Ps[wave][r][32 + l15] = (__bf16)p[2][e];
      Ps[wave][r][48 + l15] = (__bf16)p[3][e];
    }
    asm volatile("s_wait_dscnt 0" ::: "memory");
    v16bf pa[2];
#pragma unroll
    for (int c = 0; c < 2; ++c) {
      const __bf16* pp = &Ps[wave][l15][c * 32 + kbA];
      bf16x8 lo = *(const bf16x8*)pp;
      bf16x8 hi = *(const bf16x8*)(pp + 16);
#pragma unroll
      for (int i = 0; i < 8; ++i) { pa[c][i] = lo[i]; pa[c][i + 8] = hi[i]; }
    }

    // ---- O += P @ V : 2 key-chunks x 4 dh tiles ----
#pragma unroll
    for (int c = 0; c < 2; ++c) {
#pragma unroll
      for (int nt2 = 0; nt2 < 4; ++nt2) {
        v16bf bv;
        const __bf16* vp0 = &Vs[buf][nt2 * 16 + l15][c * 32 + kbB];
        bf16x8 lo = *(const bf16x8*)vp0;
        bf16x8 hi = *(const bf16x8*)(vp0 + 16);
#pragma unroll
        for (int i = 0; i < 8; ++i) { bv[i] = lo[i]; bv[i + 8] = hi[i]; }
        v8f cc;
#pragma unroll
        for (int e = 0; e < 8; ++e) cc[e] = oacc[nt2][e];
        cc = __builtin_amdgcn_wmma_f32_16x16x32_bf16(
            false, pa[c], false, bv, (short)0, cc, false, false);
#pragma unroll
        for (int e = 0; e < 8; ++e) oacc[nt2][e] = cc[e];
      }
    }
    __syncthreads();
  }

  // ---- finalize: O / l, store fp32 [b,h,n,dh] ----
  float* Op = out + (size_t)bh * NCTX * DH;
#pragma unroll
  for (int e = 0; e < 8; ++e) {
    const float inv = 1.0f / lrow[e];
    const int row = r0 + hh + e;
#pragma unroll
    for (int nt2 = 0; nt2 < 4; ++nt2)
      Op[(size_t)row * DH + nt2 * 16 + l15] = oacc[nt2][e] * inv;
  }
}

// ---------------------------------------------------------------------------
extern "C" void kernel_launch(void* const* d_in, const int* in_sizes, int n_in,
                              void* d_out, int out_size, void* d_ws,
                              size_t ws_size, hipStream_t stream) {
  (void)in_sizes; (void)n_in; (void)out_size; (void)ws_size;
  const float* x     = (const float*)d_in[0];
  const float* gamma = (const float*)d_in[1];
  const float* w_qkv = (const float*)d_in[2];
  float* out = (float*)d_out;

  // ws (bf16): xn[8192*1024] | q | k | vT (each 16*4096*64) | Wt[1536*1024]
  __bf16* xn  = (__bf16*)d_ws;
  __bf16* qb  = xn  + (size_t)ROWS * DIM;
  __bf16* kb  = qb  + (size_t)BH * NCTX * DH;
  __bf16* vtb = kb  + (size_t)BH * NCTX * DH;
  __bf16* wtb = vtb + (size_t)BH * NCTX * DH;

  rmsnorm_kernel<<<ROWS, 256, 0, stream>>>(x, gamma, xn);

  wt_transpose_kernel<<<dim3(QKV_COLS / 32, DIM / 32), 256, 0, stream>>>(w_qkv,
                                                                         wtb);

  qkv_gemm_kernel<<<dim3(QKV_COLS / 128, ROWS / 128), 256, 0, stream>>>(
      xn, wtb, qb, kb, vtb);

  flash_attn_kernel<<<dim3(NCTX / 64, BH), 128, 0, stream>>>(qb, kb, vtb, out);
}
